// LotkaVolterra_90374701842887
// MI455X (gfx1250) — compile-verified, matrix operand order checked by reading, not run on hardware
//
#include <hip/hip_runtime.h>
#include <hip/hip_bf16.h>
#include <math.h>

typedef _Float16 h16 __attribute__((ext_vector_type(16)));
typedef _Float16 h8  __attribute__((ext_vector_type(8)));
typedef float    v8f __attribute__((ext_vector_type(8)));
typedef float    f4  __attribute__((ext_vector_type(4)));

#define PNUM 4096
#define HDIM 256
#define LT   100
#define DTC  0.1f
#define SQRT_DTC 0.31622776601683794f

// ---- packed weight fragment offsets (in halfs) --------------------------
// A-fragment layout per tile: [32 lanes][16 halfs] = 512 halfs (1 KB)
// Layer0: 16 tiles (mt=0..15, kk=0, K padded 8->32)
// Layer1: 128 tiles (mt*8+kk), Layer2: 128 tiles, Layer3: 8 tiles (mt=0)
#define OFF_A0 0
#define OFF_A1 8192
#define OFF_A2 73728
#define OFF_A3 139264
#define PACK_TOTAL 143360

// d_out layout: path [P][2][101] | mu [P][100][2] | sigma [P][100][2][2]
#define MU_OFF  827392
#define SIG_OFF 1646592

__device__ __forceinline__ v8f wmma16(h16 a, h16 b, v8f c) {
    return __builtin_amdgcn_wmma_f32_16x16x32_f16(false, a, false, b,
                                                  (short)0, c, false, false);
}

__device__ __forceinline__ float softplusf(float x) {
    return fmaxf(x, 0.f) + log1pf(expf(-fabsf(x)));
}

// Pack W^T into WMMA A-fragment order (f16), zero-padded.
// A-frag 16x32 (16-bit): lane<16 holds M=lane, K={j<8 ? j : j+8}
//                        lane>=16 holds M=lane-16, K={j<8 ? j+8 : j+16}
__global__ void lv_pack_weights(const float* __restrict__ W0,
                                const float* __restrict__ W1,
                                const float* __restrict__ W2,
                                const float* __restrict__ W3,
                                _Float16* __restrict__ ws) {
    int idx = blockIdx.x * blockDim.x + threadIdx.x;
    if (idx >= PACK_TOTAL) return;

    const float* W; int Kf, Kdim, Mvalid, Mstride; int rel;
    if (idx < OFF_A1)      { W = W0; Kf = 1; Kdim = 8;   Mvalid = 256; Mstride = 256; rel = idx - OFF_A0; }
    else if (idx < OFF_A2) { W = W1; Kf = 8; Kdim = 256; Mvalid = 256; Mstride = 256; rel = idx - OFF_A1; }
    else if (idx < OFF_A3) { W = W2; Kf = 8; Kdim = 256; Mvalid = 256; Mstride = 256; rel = idx - OFF_A2; }
    else                   { W = W3; Kf = 8; Kdim = 256; Mvalid = 5;   Mstride = 5;   rel = idx - OFF_A3; }

    int j    = rel & 15;
    int lane = (rel >> 4) & 31;
    int tile = rel >> 9;
    int kk   = tile % Kf;
    int mt   = tile / Kf;

    int kl = (lane < 16) ? ((j < 8) ? j : j + 8) : ((j < 8) ? j + 8 : j + 16);
    int k  = kk * 32 + kl;
    int m  = mt * 16 + (lane & 15);

    float v = (k < Kdim && m < Mvalid) ? W[k * Mstride + m] : 0.0f;
    ws[idx] = (_Float16)v;
}

// H x H dense layer: h_out^T = relu(W^T @ h_in^T + b), both operands/results
// live in LDS in B-fragment order. One wave handles 2 N-tiles (32 particles).
__device__ __forceinline__ void dense256(const _Float16* __restrict__ wA,
                                         const float* __restrict__ bias,
                                         const _Float16* in, _Float16* outb,
                                         int lane, int hi8) {
#pragma unroll 1
    for (int mt = 0; mt < 16; ++mt) {
        v8f a0 = {0,0,0,0,0,0,0,0};
        v8f a1 = {0,0,0,0,0,0,0,0};
        if (mt < 15)
            __builtin_prefetch(wA + (mt + 1) * 8 * 512 + lane * 16, 0, 1);
#pragma unroll
        for (int kk = 0; kk < 8; ++kk) {
            h16 af  = *reinterpret_cast<const h16*>(wA + (mt * 8 + kk) * 512 + lane * 16);
            h16 bf0 = *reinterpret_cast<const h16*>(in + kk * 512 + lane * 16);
            h16 bf1 = *reinterpret_cast<const h16*>(in + (8 + kk) * 512 + lane * 16);
            a0 = wmma16(af, bf0, a0);
            a1 = wmma16(af, bf1, a1);
        }
        f4 blo = *reinterpret_cast<const f4*>(bias + mt * 16 + hi8);
        f4 bhi = *reinterpret_cast<const f4*>(bias + mt * 16 + hi8 + 4);
        h8 o0, o1;
#pragma unroll
        for (int v = 0; v < 4; ++v) {
            o0[v]     = (_Float16)fmaxf(a0[v]     + blo[v], 0.f);
            o0[v + 4] = (_Float16)fmaxf(a0[v + 4] + bhi[v], 0.f);
            o1[v]     = (_Float16)fmaxf(a1[v]     + blo[v], 0.f);
            o1[v + 4] = (_Float16)fmaxf(a1[v + 4] + bhi[v], 0.f);
        }
        // D-tile mt fills halves [(mt&1)*8, +8) of B-frag kk = mt>>1
        *reinterpret_cast<h8*>(outb + (mt >> 1) * 512 + lane * 16 + (mt & 1) * 8) = o0;
        *reinterpret_cast<h8*>(outb + (8 + (mt >> 1)) * 512 + lane * 16 + (mt & 1) * 8) = o1;
    }
}

__global__ void __launch_bounds__(64)
lv_main(const float* __restrict__ b0, const float* __restrict__ b1,
        const float* __restrict__ b2, const float* __restrict__ b3,
        const float* __restrict__ obs, const float* __restrict__ featinit,
        const float* __restrict__ tn_s, const float* __restrict__ x1_s,
        const float* __restrict__ x2_s, const float* __restrict__ seeds,
        const _Float16* __restrict__ wpack, float* __restrict__ out) {
    extern __shared__ char smem[];

    const int tid  = threadIdx.x;
    const int wave = tid >> 5;
    const int lane = tid & 31;
    const int l15  = lane & 15;
    const bool lo  = lane < 16;
    const int hi8  = ((lane >> 4) & 1) * 8;

    const int base = (blockIdx.x * 2 + wave) * 32;
    const int pA = base + l15;
    const int pB = pA + 16;

    _Float16* hb0 = reinterpret_cast<_Float16*>(smem + wave * 32768);
    _Float16* hb1 = hb0 + 8192;

    const _Float16* wA0 = wpack + OFF_A0;
    const _Float16* wA1 = wpack + OFF_A1;
    const _Float16* wA2 = wpack + OFF_A2;
    const _Float16* wA3 = wpack + OFF_A3;

    // persistent per-lane state (valid on lanes 0-15)
    float stA0 = obs[pA * 2], stA1 = obs[pA * 2 + 1];
    float stB0 = obs[pB * 2], stB1 = obs[pB * 2 + 1];
    const float t0f = featinit[0];

    float b3v[5];
#pragma unroll
    for (int v = 0; v < 5; ++v) b3v[v] = b3[v];

    // path[:, :, 0] = obs_init
    if (lo) {
        out[pA * 202 + 0]   = stA0;  out[pA * 202 + 101] = stA1;
        out[pB * 202 + 0]   = stB0;  out[pB * 202 + 101] = stB1;
    }

#pragma unroll 1
    for (int t = 0; t < LT; ++t) {
        // ---- build x^T B-fragments (K=8 live, rest zero) ----
        float fA[6], fB[6];
        if (t == 0) {
#pragma unroll
            for (int j = 0; j < 6; ++j) {
                fA[j] = featinit[pA * 6 + j];
                fB[j] = featinit[pB * 6 + j];
            }
        } else {
            float tnew = t0f + DTC * (float)t;
            float tn = tn_s[t - 1], x1 = x1_s[t - 1], x2 = x2_s[t - 1];
            fA[0] = tnew; fA[1] = tn; fA[2] = x1; fA[3] = x2; fA[4] = x1; fA[5] = x2;
#pragma unroll
            for (int j = 0; j < 6; ++j) fB[j] = fA[j];
        }
        float inA[8] = {stA0, stA1, fA[0], fA[1], fA[2], fA[3], fA[4], fA[5]};
        float inB[8] = {stB0, stB1, fB[0], fB[1], fB[2], fB[3], fB[4], fB[5]};
        h16 bx0, bx1;
#pragma unroll
        for (int j = 0; j < 8; ++j) {
            bx0[j] = lo ? (_Float16)inA[j] : (_Float16)0;
            bx1[j] = lo ? (_Float16)inB[j] : (_Float16)0;
            bx0[j + 8] = (_Float16)0;
            bx1[j + 8] = (_Float16)0;
        }

        // ---- layer 0: h0^T = relu(W0^T @ x^T + b0) -> hb0 ----
#pragma unroll 1
        for (int mt = 0; mt < 16; ++mt) {
            v8f a0 = {0,0,0,0,0,0,0,0};
            v8f a1 = {0,0,0,0,0,0,0,0};
            h16 af = *reinterpret_cast<const h16*>(wA0 + mt * 512 + lane * 16);
            a0 = wmma16(af, bx0, a0);
            a1 = wmma16(af, bx1, a1);
            f4 blo = *reinterpret_cast<const f4*>(b0 + mt * 16 + hi8);
            f4 bhi = *reinterpret_cast<const f4*>(b0 + mt * 16 + hi8 + 4);
            h8 o0, o1;
#pragma unroll
            for (int v = 0; v < 4; ++v) {
                o0[v]     = (_Float16)fmaxf(a0[v]     + blo[v], 0.f);
                o0[v + 4] = (_Float16)fmaxf(a0[v + 4] + bhi[v], 0.f);
                o1[v]     = (_Float16)fmaxf(a1[v]     + blo[v], 0.f);
                o1[v + 4] = (_Float16)fmaxf(a1[v + 4] + bhi[v], 0.f);
            }
            *reinterpret_cast<h8*>(hb0 + (mt >> 1) * 512 + lane * 16 + (mt & 1) * 8) = o0;
            *reinterpret_cast<h8*>(hb0 + (8 + (mt >> 1)) * 512 + lane * 16 + (mt & 1) * 8) = o1;
        }

        // ---- layers 1 & 2 ----
        dense256(wA1, b1, hb0, hb1, lane, hi8);
        dense256(wA2, b2, hb1, hb0, lane, hi8);

        // ---- layer 3: out^T = W3^T @ h2^T + b3 (M-tile 0 only) ----
        v8f a0 = {0,0,0,0,0,0,0,0};
        v8f a1 = {0,0,0,0,0,0,0,0};
#pragma unroll
        for (int kk = 0; kk < 8; ++kk) {
            h16 af  = *reinterpret_cast<const h16*>(wA3 + kk * 512 + lane * 16);
            h16 bf0 = *reinterpret_cast<const h16*>(hb0 + kk * 512 + lane * 16);
            h16 bf1 = *reinterpret_cast<const h16*>(hb0 + (8 + kk) * 512 + lane * 16);
            a0 = wmma16(af, bf0, a0);
            a1 = wmma16(af, bf1, a1);
        }

        // ---- sampling (lanes 0-15 hold all 5 outputs of their particle) ----
        float oA[5], oB[5];
#pragma unroll
        for (int v = 0; v < 5; ++v) { oA[v] = a0[v] + b3v[v]; oB[v] = a1[v] + b3v[v]; }

        float sA11 = softplusf(oA[2]), sA21 = oA[3], sA22 = softplusf(oA[4]);
        float sB11 = softplusf(oB[2]), sB21 = oB[3], sB22 = softplusf(oB[4]);

        float eA0 = seeds[((size_t)t * PNUM + pA) * 2];
        float eA1 = seeds[((size_t)t * PNUM + pA) * 2 + 1];
        float eB0 = seeds[((size_t)t * PNUM + pB) * 2];
        float eB1 = seeds[((size_t)t * PNUM + pB) * 2 + 1];

        float nA0 = softplusf(stA0 + DTC * oA[0] + SQRT_DTC * (sA11 * eA0));
        float nA1 = softplusf(stA1 + DTC * oA[1] + SQRT_DTC * (sA21 * eA0 + sA22 * eA1));
        float nB0 = softplusf(stB0 + DTC * oB[0] + SQRT_DTC * (sB11 * eB0));
        float nB1 = softplusf(stB1 + DTC * oB[1] + SQRT_DTC * (sB21 * eB0 + sB22 * eB1));

        if (lo) {
            size_t moA = MU_OFF + ((size_t)pA * LT + t) * 2;
            size_t moB = MU_OFF + ((size_t)pB * LT + t) * 2;
            out[moA] = oA[0]; out[moA + 1] = oA[1];
            out[moB] = oB[0]; out[moB + 1] = oB[1];

            size_t soA = SIG_OFF + ((size_t)pA * LT + t) * 4;
            size_t soB = SIG_OFF + ((size_t)pB * LT + t) * 4;
            out[soA] = sA11; out[soA + 1] = 0.f; out[soA + 2] = sA21; out[soA + 3] = sA22;
            out[soB] = sB11; out[soB + 1] = 0.f; out[soB + 2] = sB21; out[soB + 3] = sB22;

            out[pA * 202 + (t + 1)]       = nA0;
            out[pA * 202 + 101 + (t + 1)] = nA1;
            out[pB * 202 + (t + 1)]       = nB0;
            out[pB * 202 + 101 + (t + 1)] = nB1;
        }
        stA0 = nA0; stA1 = nA1; stB0 = nB0; stB1 = nB1;
    }
}

extern "C" void kernel_launch(void* const* d_in, const int* in_sizes, int n_in,
                              void* d_out, int out_size, void* d_ws, size_t ws_size,
                              hipStream_t stream) {
    const float* W0  = (const float*)d_in[0];
    const float* b0  = (const float*)d_in[1];
    const float* W1  = (const float*)d_in[2];
    const float* b1  = (const float*)d_in[3];
    const float* W2  = (const float*)d_in[4];
    const float* b2  = (const float*)d_in[5];
    const float* W3  = (const float*)d_in[6];
    const float* b3  = (const float*)d_in[7];
    const float* obs = (const float*)d_in[8];
    const float* fin = (const float*)d_in[9];
    const float* tns = (const float*)d_in[10];
    const float* x1s = (const float*)d_in[11];
    const float* x2s = (const float*)d_in[12];
    const float* sds = (const float*)d_in[13];

    _Float16* wpack = (_Float16*)d_ws;

    lv_pack_weights<<<(PACK_TOTAL + 255) / 256, 256, 0, stream>>>(W0, W1, W2, W3, wpack);

    // 64 blocks x 64 threads (2 waves), each wave owns 32 particles for the
    // entire 100-step trajectory; 64 KB dynamic LDS (2 x 32 KB wave-private
    // ping-pong activation buffers, no barriers needed).
    lv_main<<<PNUM / 64, 64, 65536, stream>>>(b0, b1, b2, b3, obs, fin, tns,
                                              x1s, x2s, sds, wpack,
                                              (float*)d_out);
}
